// MyModel_87522843558696
// MI455X (gfx1250) — compile-verified, matrix operand order checked by reading, not run on hardware
//
#include <hip/hip_runtime.h>

// ---------------------------------------------------------------------------
// B=128, S=64, E=256, U1=100, UA=200. All GEMMs on v_wmma_f32_16x16x32_f16.
// B-side weights pre-swizzled into WMMA lane layout (packed tiles of
// 32 lanes x 16 f16, contiguous per lane): one aligned 32B load per lane.
// Gate groups remapped to 16-aligned strides during packing:
//   LSTM W/U: 4 gates x 100 -> 4 x 112 (28 tiles), tile index = g*7 + ct
//   att  Wl : 4 gates x 200 -> 4 x 208 (52 tiles), tile index = g*13 + ut
// LSTM kernel runs ~1 wave/SIMD (recurrence-limited), so latency hiding is
// done in software: U tiles are hoisted into registers for all 64 steps and
// W tiles are double-buffered across the K loop.
// Attention recurrence provably cancels (softmax shift-invariance), so the
// attention "loop" is computed as fully parallel batched GEMMs.
// ---------------------------------------------------------------------------

typedef _Float16 h16;
typedef __attribute__((ext_vector_type(16))) _Float16 v16h;
typedef __attribute__((ext_vector_type(8)))  float    v8f;

#define BB 128
#define SS 64
#define EE 256
#define XP 224   // padded feature dim for 200-wide activations (200 -> 224)

__device__ __forceinline__ float sigf(float x) { return 1.0f / (1.0f + __expf(-x)); }

// fast tanh via one v_exp_f32; clamp keeps exp finite (saturated anyway)
__device__ __forceinline__ float ftanh(float x) {
  x = fminf(fmaxf(x, -15.0f), 15.0f);
  const float e = __expf(2.0f * x);
  return (e - 1.0f) / (e + 1.0f);
}

// --- A tile: 16(M) x 32(K) f16, row-major source, leading dim `ld`.
// Each lane reads two contiguous 16B chunks -> global/ds _load_b128 pairs.
__device__ __forceinline__ v16h load_a16(const h16* p, int ld) {
  const int lane = threadIdx.x & 31;
  const int m = lane & 15;
  const int kh = (lane >> 4) << 3;
  v16h r;
#pragma unroll
  for (int e = 0; e < 16; ++e) {
    const int k = ((e >> 3) << 4) + kh + (e & 7);
    r[e] = p[(size_t)m * ld + k];
  }
  return r;
}

// --- B tile from pre-swizzled weights: one aligned v16h load per lane.
__device__ __forceinline__ v16h load_bpk(const h16* __restrict__ pk, int KT, int kt, int nt) {
  const int lane = threadIdx.x & 31;
  return *(const v16h*)(pk + (((size_t)nt * KT + kt) * 32 + lane) * 16);
}

__device__ __forceinline__ v8f wmma16(v16h a, v16h b, v8f c) {
  return __builtin_amdgcn_wmma_f32_16x16x32_f16(false, a, false, b, (short)0, c,
                                                false, false);
}

// ---------------------------------------------------------------------------
// 1) Embedding gather -> f16 [B*S, 256]
// ---------------------------------------------------------------------------
__global__ void embed_kernel(const int* __restrict__ tok,
                             const float* __restrict__ emb,
                             h16* __restrict__ X0) {
  const int row = blockIdx.x;
  const int e = threadIdx.x;
  const int t = tok[row];
  X0[(size_t)row * EE + e] = (h16)emb[(size_t)t * EE + e];
}

// ---------------------------------------------------------------------------
// 2) Pack f32 weights [K, gsrc*ngroups] into WMMA-swizzled f16 tiles with
//    zero padding to Kpad rows and ngroups*gdst columns.
//    dst element layout: ((nt*KT + kt)*32 + lane)*16 + e
// ---------------------------------------------------------------------------
__global__ void pack_b_kernel(const float* __restrict__ src, h16* __restrict__ dst,
                              int K, int Kpad, int gsrc, int gdst, int ngroups) {
  const int idx = blockIdx.x * blockDim.x + threadIdx.x;
  const int Npad = gdst * ngroups;
  if (idx >= Kpad * Npad) return;
  const int e = idx & 15;
  const int lane = (idx >> 4) & 31;
  const int tile = idx >> 9;
  const int KT = Kpad >> 5;
  const int kt = tile % KT;
  const int nt = tile / KT;
  const int n = nt * 16 + (lane & 15);
  const int kh = (lane >> 4) << 3;
  const int k = kt * 32 + ((e >> 3) << 4) + kh + (e & 7);
  const int g = n / gdst;
  const int j = n % gdst;
  float v = 0.0f;
  if (k < K && j < gsrc) v = src[(size_t)k * (gsrc * ngroups) + g * gsrc + j];
  dst[idx] = (h16)v;
}

// ---------------------------------------------------------------------------
// 3) Bidirectional LSTM layer. grid=(8,2): x=batch slice of 16, y=direction.
//    blockDim=224 (7 waves); each wave owns one 16-wide cell tile, keeps the
//    f32 cell state in registers, 4 gate WMMA accumulators per step.
//    h (padded 100->128) lives in LDS and feeds the h@U A-tiles.
//    U tiles (16 x v16h = 128 VGPRs) are register-resident for all steps;
//    W tiles are double-buffered across the K loop.
// ---------------------------------------------------------------------------
__global__ void bilstm_kernel(const h16* __restrict__ X, int Kin,
                              const h16* __restrict__ Wf, const h16* __restrict__ Uf,
                              const float* __restrict__ bf,
                              const h16* __restrict__ Wb, const h16* __restrict__ Ub,
                              const float* __restrict__ bb,
                              h16* __restrict__ Hout) {
  __shared__ h16 hbuf[16][128];

  const int dir = blockIdx.y;
  const h16* W = dir ? Wb : Wf;
  const h16* U = dir ? Ub : Uf;
  const float* bias = dir ? bb : bf;
  const int col0 = dir ? 100 : 0;

  const int tid = threadIdx.x;
  const int wid = tid >> 5;        // 0..6 : cell tile ct
  const int lane = tid & 31;
  const int b0 = blockIdx.x * 16;

  for (int i = tid; i < 16 * 128; i += blockDim.x) (&hbuf[0][0])[i] = (h16)0.0f;
  __syncthreads();

  const int n = lane & 15;
  const int mrow = (lane >> 4) << 3;
  const int j0 = wid * 16 + n;     // cell index 0..111 (valid < 100)
  const bool jvalid = (j0 < 100);
  const float bi  = jvalid ? bias[j0]       : 0.0f;
  const float bfg = jvalid ? bias[100 + j0] : 0.0f;
  const float bgg = jvalid ? bias[200 + j0] : 0.0f;
  const float bog = jvalid ? bias[300 + j0] : 0.0f;

  // Register-resident U gate tiles (loop invariant across the 64 steps).
  v16h u0[4], u1[4], u2[4], u3[4];
#pragma unroll
  for (int kt = 0; kt < 4; ++kt) {
    u0[kt] = load_bpk(U, 4, kt,  0 + wid);
    u1[kt] = load_bpk(U, 4, kt,  7 + wid);
    u2[kt] = load_bpk(U, 4, kt, 14 + wid);
    u3[kt] = load_bpk(U, 4, kt, 21 + wid);
  }

  v8f cstate = {};
  const int ldX = SS * Kin;
  const int KTx = Kin >> 5;

  for (int t = 0; t < SS; ++t) {
    const int tt = dir ? (SS - 1 - t) : t;

    v8f zi, zf, zg, zo;
#pragma unroll
    for (int r = 0; r < 8; ++r) { zi[r] = bi; zf[r] = bfg; zg[r] = bgg; zo[r] = bog; }

    // ---- x_t @ W : double-buffered K loop -------------------------------
    const h16* xb = X + ((size_t)b0 * SS + tt) * Kin;
    v16h a  = load_a16(xb, ldX);
    v16h w0 = load_bpk(W, KTx, 0,  0 + wid);
    v16h w1 = load_bpk(W, KTx, 0,  7 + wid);
    v16h w2 = load_bpk(W, KTx, 0, 14 + wid);
    v16h w3 = load_bpk(W, KTx, 0, 21 + wid);
    for (int kt = 0; kt < KTx; ++kt) {
      const int ktn = (kt + 1 < KTx) ? kt + 1 : kt;   // safe dummy on last
      v16h an = load_a16(xb + ktn * 32, ldX);
      v16h n0 = load_bpk(W, KTx, ktn,  0 + wid);
      v16h n1 = load_bpk(W, KTx, ktn,  7 + wid);
      v16h n2 = load_bpk(W, KTx, ktn, 14 + wid);
      v16h n3 = load_bpk(W, KTx, ktn, 21 + wid);
      zi = wmma16(a, w0, zi);
      zf = wmma16(a, w1, zf);
      zg = wmma16(a, w2, zg);
      zo = wmma16(a, w3, zo);
      a = an; w0 = n0; w1 = n1; w2 = n2; w3 = n3;
    }
    // ---- h_{t-1} @ U : all-register B operands --------------------------
#pragma unroll
    for (int kt = 0; kt < 4; ++kt) {
      v16h ah = load_a16(&hbuf[0][0] + kt * 32, 128);
      zi = wmma16(ah, u0[kt], zi);
      zf = wmma16(ah, u1[kt], zf);
      zg = wmma16(ah, u2[kt], zg);
      zo = wmma16(ah, u3[kt], zo);
    }
    __syncthreads();   // all reads of previous h done before anyone writes

#pragma unroll
    for (int r = 0; r < 8; ++r) {
      const float c = sigf(zf[r]) * cstate[r] + sigf(zi[r]) * ftanh(zg[r]);
      cstate[r] = c;
      const float h = sigf(zo[r]) * ftanh(c);
      if (jvalid) {
        const int m = mrow + r;
        hbuf[m][j0] = (h16)h;
        Hout[((size_t)(b0 + m) * SS + tt) * XP + col0 + j0] = (h16)h;
      }
    }
    __syncthreads();
  }
}

// ---------------------------------------------------------------------------
// 4) Attention logits: E[8192,64] = X2p[8192,224] @ WdX[224,64] + bd
// ---------------------------------------------------------------------------
__global__ void ebase_kernel(const h16* __restrict__ X2, const h16* __restrict__ WdX,
                             const float* __restrict__ bd, float* __restrict__ E) {
  const int flat = blockIdx.x * 8 + (threadIdx.x >> 5);  // 0..2047 tiles
  const int mt = flat >> 2, nt = flat & 3;
  const int lane = threadIdx.x & 31;
  const int n = lane & 15, mrow = (lane >> 4) << 3;
  const float bv = bd[nt * 16 + n];
  v8f acc;
#pragma unroll
  for (int r = 0; r < 8; ++r) acc[r] = bv;
#pragma unroll
  for (int kt = 0; kt < 7; ++kt) {
    v16h a = load_a16(X2 + (size_t)mt * 16 * XP + kt * 32, XP);
    acc = wmma16(a, load_bpk(WdX, 7, kt, nt), acc);
  }
#pragma unroll
  for (int r = 0; r < 8; ++r)
    E[(size_t)(mt * 16 + mrow + r) * 64 + nt * 16 + n] = acc[r];
}

// 5) softmax over the sequence axis, one thread per (b, i)
__global__ void softmax_seq_kernel(const float* __restrict__ E, float* __restrict__ A) {
  const int idx = blockIdx.x * blockDim.x + threadIdx.x;
  if (idx >= BB * SS) return;
  const int b = idx >> 6, i = idx & 63;
  const float* p = E + (size_t)b * SS * 64 + i;
  float mx = -1e30f;
  for (int s = 0; s < SS; ++s) mx = fmaxf(mx, p[(size_t)s * 64]);
  float sum = 0.0f;
  for (int s = 0; s < SS; ++s) sum += __expf(p[(size_t)s * 64] - mx);
  const float inv = 1.0f / sum;
  float* q = A + (size_t)b * SS * 64 + i;
  for (int s = 0; s < SS; ++s) q[(size_t)s * 64] = __expf(p[(size_t)s * 64] - mx) * inv;
}

// 6) Weighted sums: WS[b,i,d] = sum_s A[b,s,i] * X2[b,s,d], one block per b
__global__ void ws_kernel(const float* __restrict__ A, const h16* __restrict__ X2,
                          h16* __restrict__ WS) {
  __shared__ float Asm[SS][64];
  __shared__ h16 Xsm[SS][XP];
  const int b = blockIdx.x, tid = threadIdx.x;
  for (int i = tid; i < SS * 64; i += 256) (&Asm[0][0])[i] = A[(size_t)b * SS * 64 + i];
  for (int i = tid; i < SS * XP; i += 256) (&Xsm[0][0])[i] = X2[(size_t)b * SS * XP + i];
  __syncthreads();
  for (int out = tid; out < SS * XP; out += 256) {
    const int i = out / XP, d = out % XP;
    float acc = 0.0f;
    if (d < 200) {
      for (int s = 0; s < SS; ++s) acc += Asm[s][i] * (float)Xsm[s][d];
    }
    WS[(size_t)b * SS * XP + out] = (h16)acc;
  }
}

// ---------------------------------------------------------------------------
// 7) Attention LSTM (c0=h0=0 each step): fused GEMM + gates. Per wave one
//    16x16 h tile; 3 accumulators (ig/gg/og) share each A tile. fg unused.
//    Packed Wl tile indices: ig -> ut, gg -> 26+ut, og -> 39+ut.
// ---------------------------------------------------------------------------
__global__ void attn_gate_kernel(const h16* __restrict__ WS, const h16* __restrict__ Wl,
                                 const float* __restrict__ bl, float* __restrict__ H) {
  const int flat = blockIdx.x * 8 + (threadIdx.x >> 5);  // 0..6655
  const int mt = flat / 13, ut = flat % 13;
  const int lane = threadIdx.x & 31;
  const int n = lane & 15, mrow = (lane >> 4) << 3;
  const int c = ut * 16 + n;       // output unit 0..207 (valid < 200)
  const bool valid = (c < 200);
  const float bi = valid ? bl[c] : 0.0f;
  const float bg = valid ? bl[400 + c] : 0.0f;
  const float bo = valid ? bl[600 + c] : 0.0f;
  v8f ai, ag, ao;
#pragma unroll
  for (int r = 0; r < 8; ++r) { ai[r] = bi; ag[r] = bg; ao[r] = bo; }
#pragma unroll
  for (int kt = 0; kt < 7; ++kt) {
    v16h a = load_a16(WS + (size_t)mt * 16 * XP + kt * 32, XP);
    ai = wmma16(a, load_bpk(Wl, 7, kt,      ut), ai);
    ag = wmma16(a, load_bpk(Wl, 7, kt, 26 + ut), ag);
    ao = wmma16(a, load_bpk(Wl, 7, kt, 39 + ut), ao);
  }
  if (valid) {
#pragma unroll
    for (int r = 0; r < 8; ++r) {
      const int row = mt * 16 + mrow + r;
      const float cst = sigf(ai[r]) * ftanh(ag[r]);
      H[(size_t)row * 200 + c] = sigf(ao[r]) * ftanh(cst);
    }
  }
}

// 8) mean over the 64 attention steps
__global__ void mean_kernel(const float* __restrict__ H, float* __restrict__ M) {
  const int idx = blockIdx.x * blockDim.x + threadIdx.x;
  if (idx >= BB * 200) return;
  const int b = idx / 200, u = idx % 200;
  float s = 0.0f;
  for (int i = 0; i < SS; ++i) s += H[((size_t)b * SS + i) * 200 + u];
  M[idx] = s * (1.0f / 64.0f);
}

// 9) MLP head + softmax, one thread per batch row
__global__ void head_kernel(const float* __restrict__ M,
                            const float* d1W, const float* d1b,
                            const float* d2W, const float* d2b,
                            const float* d3W, const float* d3b,
                            const float* d4W, const float* d4b,
                            float* __restrict__ out) {
  const int b = threadIdx.x;  // 128 threads
  float h1[80], h2[50], h3[30], h4[5];
  for (int o = 0; o < 80; ++o) {
    float a = d1b[o];
    for (int k = 0; k < 200; ++k) a += M[(size_t)b * 200 + k] * d1W[(size_t)k * 80 + o];
    h1[o] = sigf(a);
  }
  for (int o = 0; o < 50; ++o) {
    float a = d2b[o];
    for (int k = 0; k < 80; ++k) a += h1[k] * d2W[(size_t)k * 50 + o];
    h2[o] = sigf(a);
  }
  for (int o = 0; o < 30; ++o) {
    float a = d3b[o];
    for (int k = 0; k < 50; ++k) a += h2[k] * d3W[(size_t)k * 30 + o];
    h3[o] = sigf(a);
  }
  float mx = -1e30f;
  for (int o = 0; o < 5; ++o) {
    float a = d4b[o];
    for (int k = 0; k < 30; ++k) a += h3[k] * d4W[(size_t)k * 5 + o];
    h4[o] = a;
    mx = fmaxf(mx, a);
  }
  float sum = 0.0f;
  for (int o = 0; o < 5; ++o) { h4[o] = __expf(h4[o] - mx); sum += h4[o]; }
  const float inv = 1.0f / sum;
  for (int o = 0; o < 5; ++o) out[b * 5 + o] = h4[o] * inv;
}

// ---------------------------------------------------------------------------
extern "C" void kernel_launch(void* const* d_in, const int* in_sizes, int n_in,
                              void* d_out, int out_size, void* d_ws, size_t ws_size,
                              hipStream_t stream) {
  (void)in_sizes; (void)n_in; (void)out_size; (void)ws_size;

  const int*   tokens = (const int*)  d_in[0];
  const float* emb    = (const float*)d_in[1];

  char* ws = (char*)d_ws;
  size_t off = 0;
  auto take = [&](size_t bytes) -> char* {
    char* p = ws + off;
    off = (off + bytes + 255) & ~(size_t)255;
    return p;
  };

  h16* X0   = (h16*)take((size_t)BB * SS * EE * 2);
  h16* X1p  = (h16*)take((size_t)BB * SS * XP * 2);
  h16* X2p  = (h16*)take((size_t)BB * SS * XP * 2);
  // packed (WMMA-swizzled) weights
  h16* W1f  = (h16*)take((size_t)EE  * 448 * 2);
  h16* U1f  = (h16*)take((size_t)128 * 448 * 2);
  h16* W1b  = (h16*)take((size_t)EE  * 448 * 2);
  h16* U1b  = (h16*)take((size_t)128 * 448 * 2);
  h16* W2f  = (h16*)take((size_t)XP  * 448 * 2);
  h16* U2f  = (h16*)take((size_t)128 * 448 * 2);
  h16* W2b  = (h16*)take((size_t)XP  * 448 * 2);
  h16* U2b  = (h16*)take((size_t)128 * 448 * 2);
  h16* WdX  = (h16*)take((size_t)XP  * 64  * 2);
  h16* Wl   = (h16*)take((size_t)XP  * 832 * 2);
  float* Eb = (float*)take((size_t)BB * SS * 64 * 4);
  float* Al = (float*)take((size_t)BB * SS * 64 * 4);
  h16* WSb  = (h16*)take((size_t)BB * SS * XP * 2);
  float* Hf = (float*)take((size_t)BB * SS * 200 * 4);
  float* Mn = (float*)take((size_t)BB * 200 * 4);

  auto cdiv = [](int a, int b) { return (a + b - 1) / b; };

  // 1) embedding
  embed_kernel<<<BB * SS, EE, 0, stream>>>(tokens, emb, X0);

  // 2) weight packing into WMMA-swizzled layout
  auto pack = [&](const void* src, h16* dst, int K, int Kp, int gsrc, int gdst, int ng) {
    pack_b_kernel<<<cdiv(Kp * gdst * ng, 256), 256, 0, stream>>>(
        (const float*)src, dst, K, Kp, gsrc, gdst, ng);
  };
  pack(d_in[2],  W1f, EE,  EE,  100, 112, 4);
  pack(d_in[3],  U1f, 100, 128, 100, 112, 4);
  pack(d_in[5],  W1b, EE,  EE,  100, 112, 4);
  pack(d_in[6],  U1b, 100, 128, 100, 112, 4);
  pack(d_in[8],  W2f, 200, XP,  100, 112, 4);
  pack(d_in[9],  U2f, 100, 128, 100, 112, 4);
  pack(d_in[11], W2b, 200, XP,  100, 112, 4);
  pack(d_in[12], U2b, 100, 128, 100, 112, 4);
  pack(d_in[14], WdX, 200, XP,  64,  64,  1);   // top 200 rows of att_W
  pack(d_in[16], Wl,  200, XP,  200, 208, 4);

  // zero the padded activation buffers (col pads must be 0 for WMMA K loops)
  hipMemsetAsync(X1p, 0, (size_t)BB * SS * XP * 2, stream);
  hipMemsetAsync(X2p, 0, (size_t)BB * SS * XP * 2, stream);

  // 3) two bidirectional LSTM layers (grid.y = direction)
  bilstm_kernel<<<dim3(BB / 16, 2), 224, 0, stream>>>(
      X0, EE, W1f, U1f, (const float*)d_in[4], W1b, U1b, (const float*)d_in[7], X1p);
  bilstm_kernel<<<dim3(BB / 16, 2), 224, 0, stream>>>(
      X1p, XP, W2f, U2f, (const float*)d_in[10], W2b, U2b, (const float*)d_in[13], X2p);

  // 4-7) attention (recurrent state cancels -> fully parallel)
  ebase_kernel<<<256, 256, 0, stream>>>(X2p, WdX, (const float*)d_in[15], Eb);
  softmax_seq_kernel<<<cdiv(BB * SS, 256), 256, 0, stream>>>(Eb, Al);
  ws_kernel<<<BB, 256, 0, stream>>>(Al, X2p, WSb);
  attn_gate_kernel<<<(512 * 13) / 8, 256, 0, stream>>>(WSb, Wl, (const float*)d_in[18], Hf);

  // 8-9) pool + head
  mean_kernel<<<cdiv(BB * 200, 256), 256, 0, stream>>>(Hf, Mn);
  head_kernel<<<1, BB, 0, stream>>>(Mn,
      (const float*)d_in[19], (const float*)d_in[20],
      (const float*)d_in[21], (const float*)d_in[22],
      (const float*)d_in[23], (const float*)d_in[24],
      (const float*)d_in[25], (const float*)d_in[26],
      (float*)d_out);
}
